// AttentionBlock_14413910245683
// MI455X (gfx1250) — matrix operand order repaired and checked by reading, not compiled, not run on hardware
//
#include <hip/hip_runtime.h>

// ---------------- CDNA5 WMMA / TDM types ----------------
typedef __attribute__((ext_vector_type(16))) __bf16   v16bf;
typedef __attribute__((ext_vector_type(8)))  __bf16   v8bf;
typedef __attribute__((ext_vector_type(2)))  __bf16   v2bf;
typedef __attribute__((ext_vector_type(8)))  float    v8f;
typedef __attribute__((ext_vector_type(4)))  unsigned v4u;
typedef __attribute__((ext_vector_type(8)))  int      v8i;
typedef __attribute__((ext_vector_type(4)))  int      v4i;

#define B_DIM 4
#define C_DIM 256
#define D_DIM 32      // head dim == one WMMA K step
#define N_DIM 4096    // H*W
#define PADV  40      // LDS row pitch (bf16): 64B data + 16B pad -> 20 DW stride
#define XPITCH 264    // proj LDS pitch: 528B = 132 DW stride, conflict-free

union U16 { v16bf v; v8bf h[2]; unsigned u[8]; };
union U8  { v8bf  h; unsigned u[4]; };

// f32 -> bf16 through native cast (backend picks v_cvt if the HW has it)
__device__ __forceinline__ __bf16 f2bf(float f) { return (__bf16)f; }
__device__ __forceinline__ unsigned pack_bf16x2(float a, float b) {
#if __has_builtin(__builtin_amdgcn_cvt_pk_bf16_f32)
    v2bf t = __builtin_amdgcn_cvt_pk_bf16_f32(a, b);
#else
    v2bf t = { (__bf16)a, (__bf16)b };
#endif
    return __builtin_bit_cast(unsigned, t);
}

// =====================================================================
// Tensor Data Mover (CDNA5 ISA): 2D tile, 2-byte
// elems, pad 4 DW after every 16 DW so a 64B tile row lands on an 80B
// LDS pitch. 6-arg builtin on this toolchain.
// =====================================================================
#if __has_builtin(__builtin_amdgcn_tensor_load_to_lds)
#define USE_TDM 1
__device__ __forceinline__ void tdm_load_2d(unsigned lds_off, const void* gaddr,
                                            unsigned tensor_d0, unsigned tensor_d1,
                                            unsigned tile_d0, unsigned tile_d1,
                                            unsigned long long stride0_elems)
{
    union { v4u v; unsigned long long q[2]; } g0 = {};
    g0.q[0] = 1ull | ((unsigned long long)lds_off << 32);            // count=1 | lds_addr
    g0.q[1] = ((unsigned long long)(uintptr_t)gaddr & ((1ull << 57) - 1))
            | (2ull << 62);                                           // global_addr | type=2
    union { v8i v; unsigned long long q[4]; } g1 = {};
    g1.q[0] = (1ull << 16)                    // data_size = 2B
            | (1ull << 20)                    // pad_enable
            | (3ull << 22)                    // pad_interval: 16 DWORDs
            | (3ull << 25)                    // pad_amount:   4 DWORDs
            | ((unsigned long long)(tensor_d0 & 0xFFFFu) << 48);
    g1.q[1] = ((unsigned long long)tensor_d0 >> 16)
            | (((unsigned long long)tensor_d1 & 0xFFFFFFFFull) << 16)
            | ((unsigned long long)(tile_d0 & 0xFFFFu) << 48);
    g1.q[2] = (unsigned long long)(tile_d1 & 0xFFFFu)
            | ((stride0_elems & 0xFFFFFFFFull) << 32);
    g1.q[3] = (stride0_elems >> 32) & 0xFFFFull;
    v4i z4 = {};
    v8i z8 = {};
    __builtin_amdgcn_tensor_load_to_lds(g0.v, g1.v, z4, z4, z8, 0);
}
#endif

// =====================================================================
// Prep 1: weights -> bf16, pre-swizzled into the exact WMMA A-operand
// register image: dst[((tile*8 + kstep)*32 + lane)*16 + e], so the GEMM
// loads one aligned v16bf per lane with zero conversion work.
//   row = tile*16 + lane%16 ; c = kstep*32 + (e<8 ? 8*hi+e : 8+8*hi+e)
// =====================================================================
__global__ __launch_bounds__(256) void prep_w(const float* __restrict__ Wq,
                                              const float* __restrict__ Wk,
                                              const float* __restrict__ Wv,
                                              __bf16* __restrict__ wpq,
                                              __bf16* __restrict__ wpk,
                                              __bf16* __restrict__ wpv)
{
    const int idx = blockIdx.x * 256 + threadIdx.x;   // 81920 total
    const float* W; __bf16* dst; int loc;
    if (idx < 8192)        { W = Wq; dst = wpq; loc = idx; }
    else if (idx < 16384)  { W = Wk; dst = wpk; loc = idx - 8192; }
    else                   { W = Wv; dst = wpv; loc = idx - 16384; }

    const int t    = loc >> 12;          // /4096: row tile
    const int sub  = loc & 4095;
    const int ks   = sub >> 9;           // /512: k step
    const int lane = (sub >> 4) & 31;
    const int e    = sub & 15;
    const int hi   = lane >> 4, l16 = lane & 15;
    const int row  = t * 16 + l16;
    const int c    = ks * 32 + ((e < 8) ? (8 * hi + e) : (8 + 8 * hi + e));
    dst[loc] = f2bf(W[row * C_DIM + c]);
}

// =====================================================================
// Prep 2: x (f32 [B][C][N]) -> bf16 same layout; pure streaming pass.
// =====================================================================
__global__ __launch_bounds__(256) void prep_x(const float* __restrict__ x,
                                              __bf16* __restrict__ xbf)
{
    const long base = ((long)blockIdx.x * 256 + threadIdx.x) * 8;
    const float* src = x + base;
    U8 o8;
#pragma unroll
    for (int p = 0; p < 4; ++p) o8.u[p] = pack_bf16x2(src[2 * p], src[2 * p + 1]);
    *(v8bf*)(xbf + base) = o8.h;
}

// =====================================================================
// Kernel 1: q/k/v projections. One 320-thread block per (b, 16-pixel
// n-tile): stage bf16 x tile transposed in LDS once; 10 waves x 2 WMMA
// row-tiles, A operands direct from pre-swizzled weights.
// =====================================================================
__global__ __launch_bounds__(320) void proj_kernel(
        const __bf16* __restrict__ xbf,
        const __bf16* __restrict__ wpq, const float* __restrict__ bq,
        const __bf16* __restrict__ wpk, const float* __restrict__ bk,
        const __bf16* __restrict__ wpv, const float* __restrict__ bv,
        __bf16* __restrict__ qws, __bf16* __restrict__ kws,
        __bf16* __restrict__ vws)
{
    __shared__ __bf16 xT[16 * XPITCH];    // [pixel][channel]

    const int tid  = threadIdx.x;
    const int lane = tid & 31;
    const int wid  = tid >> 5;
    const int hi   = lane >> 4;
    const int l16  = lane & 15;
    const int b    = blockIdx.x >> 8;
    const int n0   = (blockIdx.x & 255) * 16;

    // transpose-stage: one channel per thread, 16 pixels
    for (int c = tid; c < C_DIM; c += 320) {
        const __bf16* src = xbf + ((long)b * C_DIM + c) * N_DIM + n0;
        const v8bf a0 = *(const v8bf*)src;
        const v8bf a1 = *(const v8bf*)(src + 8);
#pragma unroll
        for (int e = 0; e < 8; ++e) {
            xT[e * XPITCH + c]       = a0[e];
            xT[(8 + e) * XPITCH + c] = a1[e];
        }
    }
    __syncthreads();

    const int uw = __builtin_amdgcn_readfirstlane(wid);
#pragma unroll
    for (int tt = 0; tt < 2; ++tt) {
        const int t = uw * 2 + tt;        // 0,1=q  2,3=k  4..19=v
        const __bf16* Wp; const float* bias; int rowTile; int vmode;
        if (t < 2)      { Wp = wpq; bias = bq; rowTile = t;     vmode = 0; }
        else if (t < 4) { Wp = wpk; bias = bk; rowTile = t - 2; vmode = 0; }
        else            { Wp = wpv; bias = bv; rowTile = t - 4; vmode = 1; }

        v8f acc = {};
        for (int ks = 0; ks < 8; ++ks) {
            const v16bf a = *(const v16bf*)(Wp + ((rowTile * 8 + ks) * 32 + lane) * 16);
            U16 bb;
            bb.h[0] = *(const v8bf*)&xT[l16 * XPITCH + ks * 32 + 8 * hi];
            bb.h[1] = *(const v8bf*)&xT[l16 * XPITCH + ks * 32 + 8 * hi + 16];
            acc = __builtin_amdgcn_wmma_f32_16x16x32_bf16(false, a, false, bb.v,
                                                          (short)0, acc, false, false);
        }
#pragma unroll
        for (int r = 0; r < 8; ++r) {
            const int row   = rowTile * 16 + r + 8 * hi;
            const float val = acc[r] + bias[row];
            if (vmode == 0) {
                __bf16* dst = (t < 2) ? qws : kws;
                dst[((long)b * N_DIM + n0 + l16) * D_DIM + row] = f2bf(val); // [B][N][32]
            } else {
                vws[((long)b * C_DIM + row) * N_DIM + n0 + l16] = f2bf(val); // [B][C][N]
            }
        }
    }
}

// =====================================================================
// Flash-attention inner step from padded LDS tiles. S^T trick keeps P
// in A-operand layout with zero cross-lane traffic.
// =====================================================================
__device__ __forceinline__ void attn_step(const __bf16* __restrict__ kbase,
                                          const __bf16* __restrict__ vbase,
                                          const v16bf& qb, v8f o[16],
                                          float& m_run, float& l_run,
                                          int hi, int l16)
{
    U16 ka0, ka1;
    ka0.h[0] = *(const v8bf*)(kbase + l16 * PADV + 16 * hi);
    ka0.h[1] = *(const v8bf*)(kbase + l16 * PADV + 16 * hi + 8);
    ka1.h[0] = *(const v8bf*)(kbase + (16 + l16) * PADV + 16 * hi);
    ka1.h[1] = *(const v8bf*)(kbase + (16 + l16) * PADV + 16 * hi + 8);

    v8f s0 = {}, s1 = {};
    s0 = __builtin_amdgcn_wmma_f32_16x16x32_bf16(false, ka0.v, false, qb, (short)0, s0, false, false);
    s1 = __builtin_amdgcn_wmma_f32_16x16x32_bf16(false, ka1.v, false, qb, (short)0, s1, false, false);
    // s0[r] = score(i=l16, j = jbase + r + 8*hi); s1: j + 16

    float mloc = -1e30f;
#pragma unroll
    for (int r = 0; r < 8; ++r) { mloc = fmaxf(mloc, s0[r]); mloc = fmaxf(mloc, s1[r]); }
    mloc = fmaxf(mloc, __shfl_xor(mloc, 16, 32));
    const float m_new = fmaxf(m_run, mloc);
    const float corr  = __expf(m_run - m_new);

    float p0[8], p1[8], psum = 0.0f;
#pragma unroll
    for (int r = 0; r < 8; ++r) {
        p0[r] = __expf(s0[r] - m_new);
        p1[r] = __expf(s1[r] - m_new);
        psum += p0[r] + p1[r];
    }
    psum += __shfl_xor(psum, 16, 32);
    l_run = l_run * corr + psum;
    m_run = m_new;

#pragma unroll
    for (int r = 0; r < 8; ++r) {
        const float sc = __shfl(corr, r + 8 * hi, 32);
#pragma unroll
        for (int ct = 0; ct < 16; ++ct) o[ct][r] *= sc;
    }

    U16 pa;
#pragma unroll
    for (int e = 0; e < 4; ++e) {
        pa.u[e]     = pack_bf16x2(p0[2 * e], p0[2 * e + 1]);
        pa.u[4 + e] = pack_bf16x2(p1[2 * e], p1[2 * e + 1]);
    }

#pragma unroll
    for (int ct = 0; ct < 16; ++ct) {
        U16 vb;
        vb.h[0] = *(const v8bf*)(vbase + (ct * 16 + l16) * PADV + 8 * hi);
        vb.h[1] = *(const v8bf*)(vbase + (ct * 16 + l16) * PADV + 8 * hi + 16);
        o[ct] = __builtin_amdgcn_wmma_f32_16x16x32_bf16(false, pa.v, false, vb.v,
                                                        (short)0, o[ct], false, false);
    }
}

// =====================================================================
// Kernel 2: fused flash attention + residual. 8 waves/block share K/V
// tiles (TDM double-buffered); each wave owns one 16-query i-tile and
// all 256 output channels.
// =====================================================================
__global__ __launch_bounds__(256) void attn_kernel(const float* __restrict__ x,
                                                   const __bf16* __restrict__ qws,
                                                   const __bf16* __restrict__ kws,
                                                   const __bf16* __restrict__ vws,
                                                   float* __restrict__ out)
{
    __shared__ __bf16 kT[2][32 * PADV];     //  5 KB
    __shared__ __bf16 vT[2][256 * PADV];    // 40 KB

    const int lane = threadIdx.x & 31;
    const int wid  = threadIdx.x >> 5;
    const int hi   = lane >> 4;
    const int l16  = lane & 15;
    const int b     = blockIdx.x >> 5;                 // scalar
    const int itile = ((blockIdx.x & 31) << 3) | wid;
    const int iBase = itile * 16;
    const int uw    = __builtin_amdgcn_readfirstlane(wid);

    const v16bf qb = *(const v16bf*)(qws + ((long)b * N_DIM + iBase + l16) * D_DIM + 16 * hi);

    v8f o[16];
#pragma unroll
    for (int ct = 0; ct < 16; ++ct) o[ct] = v8f{};
    float m_run = -1e30f, l_run = 0.0f;

    const int steps = N_DIM / 32;   // 128

#ifdef USE_TDM
    if (uw == 0) {
        tdm_load_2d((unsigned)(uintptr_t)&kT[0][0], kws + ((long)b * N_DIM) * D_DIM,
                    32, 32, 32, 32, 32ull);
        tdm_load_2d((unsigned)(uintptr_t)&vT[0][0], vws + (long)b * C_DIM * N_DIM,
                    32, C_DIM, 32, C_DIM, (unsigned long long)N_DIM);
    }
    for (int t = 0; t < steps; ++t) {
        const int cur = t & 1;
        if (uw == 0) {
            if (t + 1 < steps) {
                const long j1 = (long)(t + 1) * 32;
                tdm_load_2d((unsigned)(uintptr_t)&kT[cur ^ 1][0],
                            kws + ((long)b * N_DIM + j1) * D_DIM,
                            32, 32, 32, 32, 32ull);
                tdm_load_2d((unsigned)(uintptr_t)&vT[cur ^ 1][0],
                            vws + (long)b * C_DIM * N_DIM + j1,
                            32, C_DIM, 32, C_DIM, (unsigned long long)N_DIM);
                __builtin_amdgcn_s_wait_tensorcnt(2);   // current pair complete
            } else {
                __builtin_amdgcn_s_wait_tensorcnt(0);
            }
        }
        __syncthreads();
        attn_step(&kT[cur][0], &vT[cur][0], qb, o, m_run, l_run, hi, l16);
        __syncthreads();
    }
#else
    for (int t = 0; t < steps; ++t) {
        const long j0 = (long)t * 32;
        {
            const v8bf* vg = (const v8bf*)(vws + ((long)b * C_DIM + threadIdx.x) * N_DIM + j0);
            v8bf* dstv = (v8bf*)&vT[0][threadIdx.x * PADV];
#pragma unroll
            for (int p = 0; p < 4; ++p) dstv[p] = vg[p];
            if (threadIdx.x < 32) {
                const v8bf* kg = (const v8bf*)(kws + ((long)b * N_DIM + j0 + threadIdx.x) * D_DIM);
                v8bf* dstk = (v8bf*)&kT[0][threadIdx.x * PADV];
#pragma unroll
                for (int p = 0; p < 4; ++p) dstk[p] = kg[p];
            }
        }
        __syncthreads();
        attn_step(&kT[0][0], &vT[0][0], qb, o, m_run, l_run, hi, l16);
        __syncthreads();
    }
#endif

    const float linv = 1.0f / l_run;
#pragma unroll
    for (int r = 0; r < 8; ++r) {
        const float li = __shfl(linv, r + 8 * hi, 32);
        const int i = iBase + r + 8 * hi;
#pragma unroll
        for (int ct = 0; ct < 16; ++ct) {
            const long idx = ((long)b * C_DIM + ct * 16 + l16) * N_DIM + i;
            out[idx] = o[ct][r] * li + x[idx];
        }
    }
}

// =====================================================================
extern "C" void kernel_launch(void* const* d_in, const int* in_sizes, int n_in,
                              void* d_out, int out_size, void* d_ws, size_t ws_size,
                              hipStream_t stream)
{
    const float* x  = (const float*)d_in[0];
    const float* Wq = (const float*)d_in[1];
    const float* bq = (const float*)d_in[2];
    const float* Wk = (const float*)d_in[3];
    const float* bk = (const float*)d_in[4];
    const float* Wv = (const float*)d_in[5];
    const float* bv = (const float*)d_in[6];
    float* out = (float*)d_out;

    const size_t qkSz = (size_t)B_DIM * N_DIM * D_DIM;     // 524288
    const size_t vSz  = (size_t)B_DIM * C_DIM * N_DIM;     // 4194304
    __bf16* qws = (__bf16*)d_ws;
    __bf16* kws = qws + qkSz;
    __bf16* vws = kws + qkSz;
    __bf16* xbf = vws + vSz;
    __bf16* wpq = xbf + vSz;
    __bf16* wpk = wpq + 8192;
    __bf16* wpv = wpk + 8192;     // 65536 elems; total ws ~19 MB

    prep_w<<<81920 / 256, 256, 0, stream>>>(Wq, Wk, Wv, wpq, wpk, wpv);
    prep_x<<<(int)(vSz / 8 / 256), 256, 0, stream>>>(x, xbf);

    proj_kernel<<<B_DIM * (N_DIM / 16), 320, 0, stream>>>(xbf, wpq, bq, wpk, bk, wpv, bv,
                                                          qws, kws, vws);

    attn_kernel<<<B_DIM * (N_DIM / 16) / 8, 256, 0, stream>>>(x, qws, kws, vws, out);
}
